// TransformerEmbeddingModel_30219389895232
// MI455X (gfx1250) — compile-verified, matrix operand order checked by reading, not run on hardware
//
#include <hip/hip_runtime.h>
#include <hip/hip_bf16.h>
#include <math.h>

// ---------------------------------------------------------------------------
// Model dimensions (fixed by the reference)
// ---------------------------------------------------------------------------
#define BB 4
#define SS 1024
#define HH 1024
#define NHH 16
#define DHH 64
#define II 4096
#define LL 4
#define MM (BB * SS)            // 4096 token rows

typedef __attribute__((ext_vector_type(16))) __bf16 v16bf;
typedef __attribute__((ext_vector_type(8)))  float  v8f;

static __device__ __forceinline__ int kmap(int e, int lhalf) {
    // 16-bit A/B fragment element -> K index (ISA 7.12.2, 16-bit A 16x32):
    // lanes 0-15: e0..7 -> K 0..7,  e8..15 -> K 16..23
    // lanes16-31: e0..7 -> K 8..15, e8..15 -> K 24..31
    return ((e & 8) << 1) + (lhalf << 3) + (e & 7);
}

// ds_swizzle_b32 xor-butterfly helpers (group-of-32 mode: offset = xor<<10 | 0x1F)
template <int IMM>
static __device__ __forceinline__ float swzf(float x) {
    return __int_as_float(__builtin_amdgcn_ds_swizzle(__float_as_int(x), IMM));
}
static __device__ __forceinline__ float redmax_h16(float t) {   // 16-lane halves
    t = fmaxf(t, swzf<0x041F>(t));   // xor 1
    t = fmaxf(t, swzf<0x081F>(t));   // xor 2
    t = fmaxf(t, swzf<0x101F>(t));   // xor 4
    t = fmaxf(t, swzf<0x201F>(t));   // xor 8
    return t;
}
static __device__ __forceinline__ float redsum_h16(float t) {
    t += swzf<0x041F>(t); t += swzf<0x081F>(t);
    t += swzf<0x101F>(t); t += swzf<0x201F>(t);
    return t;
}
static __device__ __forceinline__ float redsum_w32(float t) {   // full wave32
    t += swzf<0x041F>(t); t += swzf<0x081F>(t);
    t += swzf<0x101F>(t); t += swzf<0x201F>(t);
    t += swzf<0x401F>(t);                                        // xor 16
    return t;
}

// ---------------------------------------------------------------------------
// Embedding gather: x[row][:] = emb[ids[row]][:]
// ---------------------------------------------------------------------------
__global__ __launch_bounds__(256) void tx_embed_kernel(const int* __restrict__ ids,
                                                       const float* __restrict__ emb,
                                                       float* __restrict__ x) {
    const int row = blockIdx.x;
    const size_t id = (size_t)ids[row];
    const float4* src = (const float4*)(emb + id * HH);
    float4* dst = (float4*)(x + (size_t)row * HH);
    dst[threadIdx.x] = src[threadIdx.x];   // 256 threads * 4 floats = 1024 = H
}

// ---------------------------------------------------------------------------
// GEMM: C[M,N] = act(A[M,K] * W[K,N] + bias[N])   (bf16 WMMA, fp32 accumulate)
// Block tile 128x128, BK=32, 256 threads = 8 waves in 2(m) x 4(n) grid.
// Software pipelined: global loads for tile t+1 in flight during WMMAs of t.
// LDS double-buffered; B stored transposed [n][k]; rows padded (18-bank stride)
// so every fragment read is a conflict-free ds_load_b128.
// ---------------------------------------------------------------------------
template <int GELU>
__global__ __launch_bounds__(256) void tx_gemm_kernel(const float* __restrict__ A,
                                                      const float* __restrict__ W,
                                                      const float* __restrict__ bias,
                                                      float* __restrict__ C,
                                                      int K, int N) {
    __shared__ __bf16 As[2][128][36];   // [buf][m][k(+pad)]
    __shared__ __bf16 Bt[2][128][36];   // [buf][n][k(+pad)] (transposed)

    const int tid   = threadIdx.x;
    const int wave  = tid >> 5;
    const int lane  = tid & 31;
    const int lhalf = lane >> 4;
    const int l16   = lane & 15;
    const int wm    = wave >> 2;        // 0..1
    const int wn    = wave & 3;         // 0..3
    const int bm    = blockIdx.y * 128;
    const int bn    = blockIdx.x * 128;

    float4 ra[4], rb[4];

    auto issue_loads = [&](int kb) {
        #pragma unroll
        for (int i = 0; i < 4; ++i) {               // A tile: [128][8 float4]
            int idx = tid + i * 256;
            int r = idx >> 3, c4 = idx & 7;
            ra[i] = *(const float4*)(A + (size_t)(bm + r) * K + kb + c4 * 4);
        }
        #pragma unroll
        for (int i = 0; i < 4; ++i) {               // W tile: [32][32 float4]
            int idx = tid + i * 256;
            int r = idx >> 5, c4 = idx & 31;
            rb[i] = *(const float4*)(W + (size_t)(kb + r) * N + bn + c4 * 4);
        }
    };
    auto store_lds = [&](int buf) {
        #pragma unroll
        for (int i = 0; i < 4; ++i) {
            int idx = tid + i * 256;
            int r = idx >> 3, c4 = idx & 7;
            __bf16* d = &As[buf][r][c4 * 4];
            d[0] = (__bf16)ra[i].x; d[1] = (__bf16)ra[i].y;
            d[2] = (__bf16)ra[i].z; d[3] = (__bf16)ra[i].w;
        }
        #pragma unroll
        for (int i = 0; i < 4; ++i) {               // transpose: W[k][n]->Bt[n][k]
            int idx = tid + i * 256;
            int r = idx >> 5, c = (idx & 31) * 4;
            Bt[buf][c + 0][r] = (__bf16)rb[i].x;
            Bt[buf][c + 1][r] = (__bf16)rb[i].y;
            Bt[buf][c + 2][r] = (__bf16)rb[i].z;
            Bt[buf][c + 3][r] = (__bf16)rb[i].w;
        }
    };

    v8f acc[4][2] = {};

    issue_loads(0);
    store_lds(0);
    __syncthreads();

    const int nsteps = K >> 5;
    for (int s = 0; s < nsteps; ++s) {
        const int buf = s & 1;
        if (s + 1 < nsteps) issue_loads((s + 1) << 5);   // loads in flight

        v16bf bfrag[2];
        #pragma unroll
        for (int nt = 0; nt < 2; ++nt) {
            const int n = wn * 32 + nt * 16 + l16;
            #pragma unroll
            for (int e = 0; e < 16; ++e) bfrag[nt][e] = Bt[buf][n][kmap(e, lhalf)];
        }
        #pragma unroll
        for (int mt = 0; mt < 4; ++mt) {
            v16bf afrag;
            const int r = wm * 64 + mt * 16 + l16;
            #pragma unroll
            for (int e = 0; e < 16; ++e) afrag[e] = As[buf][r][kmap(e, lhalf)];
            #pragma unroll
            for (int nt = 0; nt < 2; ++nt)
                acc[mt][nt] = __builtin_amdgcn_wmma_f32_16x16x32_bf16(
                    false, afrag, false, bfrag[nt], (short)0, acc[mt][nt], false, false);
        }

        if (s + 1 < nsteps) store_lds(buf ^ 1);          // cvt+store after WMMAs
        __syncthreads();
    }

    // Epilogue: bias (+ exact GELU), write fp32
    #pragma unroll
    for (int mt = 0; mt < 4; ++mt) {
        #pragma unroll
        for (int nt = 0; nt < 2; ++nt) {
            const int col = bn + wn * 32 + nt * 16 + l16;
            const float bs = bias[col];
            #pragma unroll
            for (int v = 0; v < 8; ++v) {
                const int row = bm + wm * 64 + mt * 16 + v + 8 * lhalf;
                float val = acc[mt][nt][v] + bs;
                if (GELU) val = 0.5f * val * (1.0f + erff(val * 0.70710678118654752f));
                C[(size_t)row * N + col] = val;
            }
        }
    }
}

// ---------------------------------------------------------------------------
// Flash attention: block = (q-tile of 64 rows, head, batch), 128 thr = 4 waves.
// Each wave owns a 16-row q sub-tile. Online softmax in fp32, WMMA bf16 math.
// Double-buffered K/V staging with loads in flight over the compute.
// ---------------------------------------------------------------------------
__global__ __launch_bounds__(128) void tx_attn_kernel(const float* __restrict__ q,
                                                      const float* __restrict__ k,
                                                      const float* __restrict__ v,
                                                      const int* __restrict__ mask,
                                                      float* __restrict__ ctx) {
    __shared__ __bf16 Kt[2][32][68];    // [buf][kv][dh(+pad)]
    __shared__ __bf16 Vt[2][64][36];    // [buf][dh][kv(+pad)] (transposed)
    __shared__ float  biasL[SS];
    __shared__ __bf16 Pls[4][16][36];   // per-wave P relayout patch

    const int tid   = threadIdx.x;
    const int wave  = tid >> 5;
    const int lane  = tid & 31;
    const int lhalf = lane >> 4;
    const int l16   = lane & 15;
    const int b  = blockIdx.z;
    const int h  = blockIdx.y;
    const int qt = blockIdx.x;

    // Mask bias for all key positions (uniform loop count)
    for (int c = tid; c < SS; c += 128)
        biasL[c] = (1.0f - (float)mask[(size_t)b * SS + c]) * -10000.0f;

    float4 rk[4], rv[4];
    auto issue_kv = [&](int kvb) {
        #pragma unroll
        for (int i = 0; i < 4; ++i) {           // float4 over [32][16]
            int idx = tid + i * 128;
            int r = idx >> 4, c4 = idx & 15;
            const size_t base = ((size_t)b * SS + kvb + r) * HH + h * DHH + c4 * 4;
            rk[i] = *(const float4*)(k + base);
            rv[i] = *(const float4*)(v + base);
        }
    };
    auto store_kv = [&](int buf) {
        #pragma unroll
        for (int i = 0; i < 4; ++i) {
            int idx = tid + i * 128;
            int r = idx >> 4, c = (idx & 15) * 4;
            __bf16* dk = &Kt[buf][r][c];
            dk[0] = (__bf16)rk[i].x; dk[1] = (__bf16)rk[i].y;
            dk[2] = (__bf16)rk[i].z; dk[3] = (__bf16)rk[i].w;
            Vt[buf][c + 0][r] = (__bf16)rv[i].x;    // transpose V: [dh][kv]
            Vt[buf][c + 1][r] = (__bf16)rv[i].y;
            Vt[buf][c + 2][r] = (__bf16)rv[i].z;
            Vt[buf][c + 3][r] = (__bf16)rv[i].w;
        }
    };

    // Q fragments for this wave's 16 rows (A layout, two K-chunks over DH=64)
    const int qrow = qt * 64 + wave * 16 + l16;
    const float* qp = q + ((size_t)b * SS + qrow) * HH + h * DHH;
    v16bf qfrag[2];
    #pragma unroll
    for (int c = 0; c < 2; ++c)
        #pragma unroll
        for (int e = 0; e < 16; ++e)
            qfrag[c][e] = (__bf16)qp[c * 32 + kmap(e, lhalf)];

    float mrow[8], lrow[8];
    #pragma unroll
    for (int i = 0; i < 8; ++i) { mrow[i] = -1e30f; lrow[i] = 0.0f; }
    v8f cacc[4] = {};

    issue_kv(0);
    store_kv(0);
    __syncthreads();

    for (int s = 0; s < (SS >> 5); ++s) {
        const int buf = s & 1;
        if (s + 1 < (SS >> 5)) issue_kv((s + 1) << 5);   // loads in flight
        const int kvb = s << 5;

        // S = Q * K^T  (16 x 32 per wave)
        v8f sacc[2] = {};
        #pragma unroll
        for (int nt = 0; nt < 2; ++nt) {
            const int kvcol = nt * 16 + l16;
            #pragma unroll
            for (int c = 0; c < 2; ++c) {
                v16bf bf;
                #pragma unroll
                for (int e = 0; e < 16; ++e) bf[e] = Kt[buf][kvcol][c * 32 + kmap(e, lhalf)];
                sacc[nt] = __builtin_amdgcn_wmma_f32_16x16x32_bf16(
                    false, qfrag[c], false, bf, (short)0, sacc[nt], false, false);
            }
        }
        // scale + mask bias
        const float b0 = biasL[kvb + l16];
        const float b1 = biasL[kvb + 16 + l16];
        #pragma unroll
        for (int vv = 0; vv < 8; ++vv) {
            sacc[0][vv] = sacc[0][vv] * 0.125f + b0;
            sacc[1][vv] = sacc[1][vv] * 0.125f + b1;
        }
        // online softmax (rows live in 16-lane halves; ds_swizzle butterflies)
        float fac[8];
        #pragma unroll
        for (int vv = 0; vv < 8; ++vv) {
            float t = redmax_h16(fmaxf(sacc[0][vv], sacc[1][vv]));
            t = fmaxf(t, mrow[vv]);
            fac[vv] = expf(mrow[vv] - t);
            mrow[vv] = t;
            float p0 = expf(sacc[0][vv] - t);
            float p1 = expf(sacc[1][vv] - t);
            sacc[0][vv] = p0; sacc[1][vv] = p1;
            lrow[vv] = lrow[vv] * fac[vv] + redsum_h16(p0 + p1);
        }
        #pragma unroll
        for (int d = 0; d < 4; ++d)
            #pragma unroll
            for (int vv = 0; vv < 8; ++vv) cacc[d][vv] *= fac[vv];

        // P: C-layout regs -> LDS -> A-layout fragment
        #pragma unroll
        for (int nt = 0; nt < 2; ++nt)
            #pragma unroll
            for (int vv = 0; vv < 8; ++vv)
                Pls[wave][vv + 8 * lhalf][nt * 16 + l16] = (__bf16)sacc[nt][vv];
        __syncthreads();

        v16bf pfrag;
        #pragma unroll
        for (int e = 0; e < 16; ++e) pfrag[e] = Pls[wave][l16][kmap(e, lhalf)];

        // ctx += P * V
        #pragma unroll
        for (int d = 0; d < 4; ++d) {
            v16bf vf;
            #pragma unroll
            for (int e = 0; e < 16; ++e) vf[e] = Vt[buf][d * 16 + l16][kmap(e, lhalf)];
            cacc[d] = __builtin_amdgcn_wmma_f32_16x16x32_bf16(
                false, pfrag, false, vf, (short)0, cacc[d], false, false);
        }

        if (s + 1 < (SS >> 5)) store_kv(buf ^ 1);        // cvt+store after WMMAs
        __syncthreads();
    }

    // normalize and write ctx [B,S,H]
    #pragma unroll
    for (int d = 0; d < 4; ++d) {
        #pragma unroll
        for (int vv = 0; vv < 8; ++vv) {
            const int srow = qt * 64 + wave * 16 + vv + 8 * lhalf;
            const int dh   = d * 16 + l16;
            ctx[((size_t)b * SS + srow) * HH + h * DHH + dh] = cacc[d][vv] / lrow[vv];
        }
    }
}

// ---------------------------------------------------------------------------
// LayerNorm: out[row] = LN(in1[row] (+ in2[row])) * g + b   (row length H=1024)
// ---------------------------------------------------------------------------
__global__ __launch_bounds__(256) void tx_ln_kernel(const float* __restrict__ in1,
                                                    const float* __restrict__ in2,
                                                    const float* __restrict__ g,
                                                    const float* __restrict__ bt,
                                                    float* __restrict__ out) {
    __shared__ float red[8];
    __shared__ float stat[2];
    const int tid = threadIdx.x, lane = tid & 31, wave = tid >> 5;
    const int row = blockIdx.x;
    const float* p1 = in1 + (size_t)row * HH;
    const float* p2 = in2 ? in2 + (size_t)row * HH : nullptr;

    float vals[4];
    #pragma unroll
    for (int i = 0; i < 4; ++i) {
        int c = tid + i * 256;
        vals[i] = p1[c] + (p2 ? p2[c] : 0.0f);
    }
    float s = redsum_w32(vals[0] + vals[1] + vals[2] + vals[3]);
    if (lane == 0) red[wave] = s;
    __syncthreads();
    if (tid == 0) {
        float t = 0;
        for (int i = 0; i < 8; ++i) t += red[i];
        stat[0] = t * (1.0f / HH);
    }
    __syncthreads();
    const float mu = stat[0];
    float vs = 0;
    #pragma unroll
    for (int i = 0; i < 4; ++i) { float d = vals[i] - mu; vs += d * d; }
    vs = redsum_w32(vs);
    __syncthreads();                      // red reuse: pass-1 read done
    if (lane == 0) red[wave] = vs;
    __syncthreads();
    if (tid == 0) {
        float t = 0;
        for (int i = 0; i < 8; ++i) t += red[i];
        stat[1] = rsqrtf(t * (1.0f / HH) + 1e-5f);
    }
    __syncthreads();
    const float inv = stat[1];
    #pragma unroll
    for (int i = 0; i < 4; ++i) {
        int c = tid + i * 256;
        out[(size_t)row * HH + c] = (vals[i] - mu) * inv * g[c] + bt[c];
    }
}

// ---------------------------------------------------------------------------
// Host orchestration
// ---------------------------------------------------------------------------
extern "C" void kernel_launch(void* const* d_in, const int* in_sizes, int n_in,
                              void* d_out, int out_size, void* d_ws, size_t ws_size,
                              hipStream_t stream) {
    (void)in_sizes; (void)n_in; (void)out_size; (void)ws_size;
    const int*   ids   = (const int*)  d_in[0];
    const int*   amask = (const int*)  d_in[1];
    const float* emb   = (const float*)d_in[2];
    const float* Wq    = (const float*)d_in[3];
    const float* bq    = (const float*)d_in[4];
    const float* Wk    = (const float*)d_in[5];
    const float* bk    = (const float*)d_in[6];
    const float* Wv    = (const float*)d_in[7];
    const float* bv    = (const float*)d_in[8];
    const float* Wo    = (const float*)d_in[9];
    const float* bo    = (const float*)d_in[10];
    const float* lnag  = (const float*)d_in[11];
    const float* lnab  = (const float*)d_in[12];
    const float* Wi    = (const float*)d_in[13];
    const float* bi    = (const float*)d_in[14];
    const float* Wout  = (const float*)d_in[15];
    const float* bout  = (const float*)d_in[16];
    const float* ln1g  = (const float*)d_in[17];
    const float* ln1b  = (const float*)d_in[18];
    const float* ln2g  = (const float*)d_in[19];
    const float* ln2b  = (const float*)d_in[20];
    const float* lnfg  = (const float*)d_in[21];
    const float* lnfb  = (const float*)d_in[22];
    float* outp = (float*)d_out;

    const size_t MH = (size_t)MM * HH;          // 4M floats / 16 MB
    float* ws  = (float*)d_ws;
    float* x   = ws;                            // [M,H]
    float* qb  = ws + 1 * MH;                   // [M,H]
    float* kb_ = ws + 2 * MH;                   // [M,H]
    float* vb  = ws + 3 * MH;                   // [M,H]
    float* cb  = ws + 4 * MH;                   // [M,H] attention ctx
    float* ob  = ws + 5 * MH;                   // [M,H] o-proj / ffn out
    float* att = qb;                            // reuse q after attention
    float* h1  = qb;                            // [M,I] spans q..ctx (dead then)
    float* ffn = ob;

    const dim3 gemmH(HH / 128, MM / 128);       // N=1024
    const dim3 gemmI(II / 128, MM / 128);       // N=4096
    const dim3 attnG(SS / 64, NHH, BB);

    tx_embed_kernel<<<MM, 256, 0, stream>>>(ids, emb, x);

    for (int l = 0; l < LL; ++l) {
        const size_t oHH = (size_t)l * HH * HH, oH = (size_t)l * HH;
        const size_t oHI = (size_t)l * HH * II, oI = (size_t)l * II;
        // QKV projections
        tx_gemm_kernel<0><<<gemmH, 256, 0, stream>>>(x, Wq + oHH, bq + oH, qb,  HH, HH);
        tx_gemm_kernel<0><<<gemmH, 256, 0, stream>>>(x, Wk + oHH, bk + oH, kb_, HH, HH);
        tx_gemm_kernel<0><<<gemmH, 256, 0, stream>>>(x, Wv + oHH, bv + oH, vb,  HH, HH);
        // attention
        tx_attn_kernel<<<attnG, 128, 0, stream>>>(qb, kb_, vb, amask, cb);
        // output projection + ln_attn + residual ln1
        tx_gemm_kernel<0><<<gemmH, 256, 0, stream>>>(cb, Wo + oHH, bo + oH, ob, HH, HH);
        tx_ln_kernel<<<MM, 256, 0, stream>>>(ob, nullptr, lnag + oH, lnab + oH, att);
        tx_ln_kernel<<<MM, 256, 0, stream>>>(x,  att,     ln1g + oH, ln1b + oH, x);
        // FFN
        tx_gemm_kernel<1><<<gemmI, 256, 0, stream>>>(x,  Wi   + oHI, bi   + oI, h1,  HH, II);
        tx_gemm_kernel<0><<<gemmH, 256, 0, stream>>>(h1, Wout + oHI, bout + oH, ffn, II, HH);
        tx_ln_kernel<<<MM, 256, 0, stream>>>(x, ffn, ln2g + oH, ln2b + oH, x);
    }
    // final layernorm -> output
    tx_ln_kernel<<<MM, 256, 0, stream>>>(x, nullptr, lnfg, lnfb, outp);
}
